// SNN_89172110999565
// MI455X (gfx1250) — matrix operand order repaired and checked by reading, compiled, not run on hardware
//
#include <hip/hip_runtime.h>

// ---------------------------------------------------------------------------
// SNN forward on MI455X (gfx1250): persistent per-batch-tile kernel, v2.
//   - 1 workgroup = 32 batch rows (2 rowblocks x 16), 8 waves, 25 steps fused
//   - each B-fragment (weights, L2-resident bf16) feeds TWO wmma ops
//     -> halves the dominant L2 weight-streaming traffic vs BT=16
//   - mem1/mem2/ssum in VGPRs; cur1 (loop-invariant) parked in LDS;
//     spikes exchanged via LDS (bf16 bit patterns, exact {0,1})
//   - all matmuls via v_wmma_f32_16x16x32_bf16
// ---------------------------------------------------------------------------

#define B_ROWS   16384
#define H_DIM    1024
#define O_DIM    10
#define NSTEPS   25
#define BETA_C   0.5f
#define THR_C    1.0f
#define BT       32           // batch rows per workgroup
#define SPK_STRIDE 1032       // +16B pad: b128 LDS reads rotate across 64 banks
#define CUR_STRIDE 1040       // +64B pad: b32 reads of lanes 0-15/16-31 disjoint

typedef __attribute__((ext_vector_type(16))) __bf16        v16bf;
typedef __attribute__((ext_vector_type(8)))  float         v8f;
typedef __attribute__((ext_vector_type(4)))  unsigned int  v4u;

union Frag32B { v16bf v; v4u q[2]; };   // one WMMA A/B operand = 32 bytes/lane

__device__ __forceinline__ unsigned short f2bf(float f) {
    unsigned u = __float_as_uint(f);
    u += 0x7FFFu + ((u >> 16) & 1u);          // round-to-nearest-even
    return (unsigned short)(u >> 16);
}

// fp32 -> bf16 conversion pass (x and the four H x H weight matrices)
__global__ void snn_prep_bf16(const float* __restrict__ src,
                              unsigned short* __restrict__ dst, int n) {
    int i = blockIdx.x * blockDim.x + threadIdx.x;
    int stride = gridDim.x * blockDim.x;
    for (; i < n; i += stride) dst[i] = f2bf(src[i]);
}

__launch_bounds__(256)
__global__ void snn_persistent(
    const unsigned short* __restrict__ xbf,
    const unsigned short* __restrict__ wW1,
    const unsigned short* __restrict__ wV1,
    const unsigned short* __restrict__ wW2,
    const unsigned short* __restrict__ wV2,
    const float* __restrict__ b1,  const float* __restrict__ V1b,
    const float* __restrict__ b2,  const float* __restrict__ V2b,
    const float* __restrict__ W3,  const float* __restrict__ b3,
    float* __restrict__ out)
{
    __shared__ struct {
        unsigned short spk[2][BT * SPK_STRIDE];                  // ~132 KB
        union { float cur1[BT * CUR_STRIDE];                     // ~133 KB
                float ssum[BT * CUR_STRIDE]; } u;                // (overlay)
    } sm;

    const int tid  = threadIdx.x;
    const int wave = tid >> 5;
    const int lane = tid & 31;
    const int lh   = lane & 15;     // WMMA: N (B/C/D) or M-row (A) selector
    const int lg   = lane >> 4;     // WMMA: half-wave selector
    const int row0 = blockIdx.x * BT;
    const int nb   = wave * 128;    // this wave's output-column base

    // Per-tile bias values: C/D layout has n = nb + 16*t + lh fixed per lane.
    float b1v[8], V1bv[8], b2v[8], V2bv[8];
#pragma unroll
    for (int t = 0; t < 8; ++t) {
        const int n = nb + t * 16 + lh;
        b1v[t] = b1[n]; V1bv[t] = V1b[n]; b2v[t] = b2[n]; V2bv[t] = V2b[n];
    }

    // ---- Phase 0: cur1 = x @ W1^T + b1 for both rowblocks -> LDS ----
    {
        v8f acc[2][8];
#pragma unroll
        for (int rb = 0; rb < 2; ++rb)
#pragma unroll
            for (int t = 0; t < 8; ++t)
#pragma unroll
                for (int r = 0; r < 8; ++r) acc[rb][t][r] = 0.0f;

        const unsigned short* a0 = xbf + (size_t)(row0 + lh) * H_DIM;
        const unsigned short* a1 = xbf + (size_t)(row0 + 16 + lh) * H_DIM;
        for (int kb = 0; kb < H_DIM; kb += 32) {
            Frag32B fa0, fa1;
            const v4u* p0 = reinterpret_cast<const v4u*>(a0 + kb + lg * 8);
            const v4u* p1 = reinterpret_cast<const v4u*>(a1 + kb + lg * 8);
            fa0.q[0] = p0[0]; fa0.q[1] = p0[2];      // K: +0..7 and +16..23
            fa1.q[0] = p1[0]; fa1.q[1] = p1[2];
#pragma unroll
            for (int t = 0; t < 8; ++t) {
                Frag32B fb;
                const v4u* pb = reinterpret_cast<const v4u*>(
                    wW1 + (size_t)(nb + t * 16 + lh) * H_DIM + kb + lg * 8);
                fb.q[0] = pb[0]; fb.q[1] = pb[2];    // K: +0..7 and +16..23
                acc[0][t] = __builtin_amdgcn_wmma_f32_16x16x32_bf16(
                    false, fa0.v, false, fb.v, (short)0, acc[0][t], false, false);
                acc[1][t] = __builtin_amdgcn_wmma_f32_16x16x32_bf16(
                    false, fa1.v, false, fb.v, (short)0, acc[1][t], false, false);
            }
        }
#pragma unroll
        for (int rb = 0; rb < 2; ++rb)
#pragma unroll
            for (int t = 0; t < 8; ++t)
#pragma unroll
                for (int r = 0; r < 8; ++r)
                    sm.u.cur1[(rb * 16 + r + 8 * lg) * CUR_STRIDE +
                              (nb + t * 16 + lh)] = acc[rb][t][r] + b1v[t];
    }

    // ---- Register-resident LIF state (2 rowblocks) ----
    v8f mem1[2][8], mem2[2][8], ssum[2][8];
#pragma unroll
    for (int rb = 0; rb < 2; ++rb)
#pragma unroll
        for (int t = 0; t < 8; ++t)
#pragma unroll
            for (int r = 0; r < 8; ++r) {
                mem1[rb][t][r] = 0.f; mem2[rb][t][r] = 0.f; ssum[rb][t][r] = 0.f;
            }

    // ---- 25 time steps, fully on-chip ----
    for (int step = 0; step < NSTEPS; ++step) {
        // ===== Layer 1: rec = spk1_old @ V1w^T (zero at step 0) =====
        v8f acc[2][8];
#pragma unroll
        for (int rb = 0; rb < 2; ++rb)
#pragma unroll
            for (int t = 0; t < 8; ++t)
#pragma unroll
                for (int r = 0; r < 8; ++r) acc[rb][t][r] = 0.0f;

        if (step > 0) {
            const unsigned short* a0 = &sm.spk[0][lh * SPK_STRIDE];
            const unsigned short* a1 = &sm.spk[0][(16 + lh) * SPK_STRIDE];
            for (int kb = 0; kb < H_DIM; kb += 32) {
                Frag32B fa0, fa1;
                const v4u* p0 = reinterpret_cast<const v4u*>(a0 + kb + lg * 8);
                const v4u* p1 = reinterpret_cast<const v4u*>(a1 + kb + lg * 8);
                fa0.q[0] = p0[0]; fa0.q[1] = p0[2];
                fa1.q[0] = p1[0]; fa1.q[1] = p1[2];
#pragma unroll
                for (int t = 0; t < 8; ++t) {
                    Frag32B fb;
                    const v4u* pb = reinterpret_cast<const v4u*>(
                        wV1 + (size_t)(nb + t * 16 + lh) * H_DIM + kb + lg * 8);
                    fb.q[0] = pb[0]; fb.q[1] = pb[2];
                    acc[0][t] = __builtin_amdgcn_wmma_f32_16x16x32_bf16(
                        false, fa0.v, false, fb.v, (short)0, acc[0][t], false, false);
                    acc[1][t] = __builtin_amdgcn_wmma_f32_16x16x32_bf16(
                        false, fa1.v, false, fb.v, (short)0, acc[1][t], false, false);
                }
            }
        }
        __syncthreads();   // all waves done reading spk1_old

        // mem1/spk1 update; publish new spikes (bf16 bit pattern, no cvt)
#pragma unroll
        for (int rb = 0; rb < 2; ++rb)
#pragma unroll
            for (int t = 0; t < 8; ++t)
#pragma unroll
                for (int r = 0; r < 8; ++r) {
                    const int mrow = rb * 16 + r + 8 * lg;
                    const int ncol = nb + t * 16 + lh;
                    const float c1 = sm.u.cur1[mrow * CUR_STRIDE + ncol];
                    float m = mem1[rb][t][r];
                    const float reset = (m > THR_C) ? THR_C : 0.0f;
                    m = BETA_C * m + c1 + acc[rb][t][r] + V1bv[t] - reset;
                    mem1[rb][t][r] = m;
                    sm.spk[0][mrow * SPK_STRIDE + ncol] =
                        (m - THR_C > 0.0f) ? (unsigned short)0x3F80u
                                           : (unsigned short)0u;
                }
        __syncthreads();   // spk1_new visible

        // ===== Layer 2: cur2 = spk1_new @ W2^T (+ spk2_old @ V2w^T) =====
#pragma unroll
        for (int rb = 0; rb < 2; ++rb)
#pragma unroll
            for (int t = 0; t < 8; ++t)
#pragma unroll
                for (int r = 0; r < 8; ++r) acc[rb][t][r] = 0.0f;
        {
            const unsigned short* a0 = &sm.spk[0][lh * SPK_STRIDE];
            const unsigned short* a1 = &sm.spk[0][(16 + lh) * SPK_STRIDE];
            for (int kb = 0; kb < H_DIM; kb += 32) {
                Frag32B fa0, fa1;
                const v4u* p0 = reinterpret_cast<const v4u*>(a0 + kb + lg * 8);
                const v4u* p1 = reinterpret_cast<const v4u*>(a1 + kb + lg * 8);
                fa0.q[0] = p0[0]; fa0.q[1] = p0[2];
                fa1.q[0] = p1[0]; fa1.q[1] = p1[2];
#pragma unroll
                for (int t = 0; t < 8; ++t) {
                    Frag32B fb;
                    const v4u* pb = reinterpret_cast<const v4u*>(
                        wW2 + (size_t)(nb + t * 16 + lh) * H_DIM + kb + lg * 8);
                    fb.q[0] = pb[0]; fb.q[1] = pb[2];
                    acc[0][t] = __builtin_amdgcn_wmma_f32_16x16x32_bf16(
                        false, fa0.v, false, fb.v, (short)0, acc[0][t], false, false);
                    acc[1][t] = __builtin_amdgcn_wmma_f32_16x16x32_bf16(
                        false, fa1.v, false, fb.v, (short)0, acc[1][t], false, false);
                }
            }
        }
        if (step > 0) {
            const unsigned short* a0 = &sm.spk[1][lh * SPK_STRIDE];
            const unsigned short* a1 = &sm.spk[1][(16 + lh) * SPK_STRIDE];
            for (int kb = 0; kb < H_DIM; kb += 32) {
                Frag32B fa0, fa1;
                const v4u* p0 = reinterpret_cast<const v4u*>(a0 + kb + lg * 8);
                const v4u* p1 = reinterpret_cast<const v4u*>(a1 + kb + lg * 8);
                fa0.q[0] = p0[0]; fa0.q[1] = p0[2];
                fa1.q[0] = p1[0]; fa1.q[1] = p1[2];
#pragma unroll
                for (int t = 0; t < 8; ++t) {
                    Frag32B fb;
                    const v4u* pb = reinterpret_cast<const v4u*>(
                        wV2 + (size_t)(nb + t * 16 + lh) * H_DIM + kb + lg * 8);
                    fb.q[0] = pb[0]; fb.q[1] = pb[2];
                    acc[0][t] = __builtin_amdgcn_wmma_f32_16x16x32_bf16(
                        false, fa0.v, false, fb.v, (short)0, acc[0][t], false, false);
                    acc[1][t] = __builtin_amdgcn_wmma_f32_16x16x32_bf16(
                        false, fa1.v, false, fb.v, (short)0, acc[1][t], false, false);
                }
            }
        }
        __syncthreads();   // all waves done reading spk2_old

#pragma unroll
        for (int rb = 0; rb < 2; ++rb)
#pragma unroll
            for (int t = 0; t < 8; ++t)
#pragma unroll
                for (int r = 0; r < 8; ++r) {
                    const int mrow = rb * 16 + r + 8 * lg;
                    const int ncol = nb + t * 16 + lh;
                    float m = mem2[rb][t][r];
                    const float reset = (m > THR_C) ? THR_C : 0.0f;
                    m = BETA_C * m + acc[rb][t][r] + b2v[t] + V2bv[t] - reset;
                    mem2[rb][t][r] = m;
                    const float sp = (m - THR_C > 0.0f) ? 1.0f : 0.0f;
                    ssum[rb][t][r] += sp;
                    sm.spk[1][mrow * SPK_STRIDE + ncol] =
                        (sp != 0.0f) ? (unsigned short)0x3F80u : (unsigned short)0u;
                }
        __syncthreads();   // spk2_new visible for next step
    }

    // ---- Output: out = (ssum/25) @ W3^T + b3 (O=10, tiny) ----
#pragma unroll
    for (int rb = 0; rb < 2; ++rb)
#pragma unroll
        for (int t = 0; t < 8; ++t)
#pragma unroll
            for (int r = 0; r < 8; ++r)
                sm.u.ssum[(rb * 16 + r + 8 * lg) * CUR_STRIDE +
                          (nb + t * 16 + lh)] = ssum[rb][t][r];
    __syncthreads();

    for (int idx = tid; idx < BT * O_DIM; idx += 256) {
        const int m = idx / O_DIM, o = idx % O_DIM;
        float a = 0.0f;
        const float* w3 = W3 + (size_t)o * H_DIM;
        const float* sr = &sm.u.ssum[m * CUR_STRIDE];
        for (int h = 0; h < H_DIM; ++h) a += sr[h] * w3[h];
        out[(size_t)(row0 + m) * O_DIM + o] = a * (1.0f / NSTEPS) + b3[o];
    }
}

extern "C" void kernel_launch(void* const* d_in, const int* in_sizes, int n_in,
                              void* d_out, int out_size, void* d_ws, size_t ws_size,
                              hipStream_t stream) {
    const float* x   = (const float*)d_in[0];
    const float* W1  = (const float*)d_in[1];
    const float* b1  = (const float*)d_in[2];
    const float* V1w = (const float*)d_in[3];
    const float* V1b = (const float*)d_in[4];
    const float* W2  = (const float*)d_in[5];
    const float* b2  = (const float*)d_in[6];
    const float* V2w = (const float*)d_in[7];
    const float* V2b = (const float*)d_in[8];
    const float* W3  = (const float*)d_in[9];
    const float* b3  = (const float*)d_in[10];
    float* out = (float*)d_out;

    // Workspace: bf16 copies of x and the four H x H weights (~42 MB total)
    unsigned short* xbf = (unsigned short*)d_ws;
    unsigned short* wW1 = xbf + (size_t)B_ROWS * H_DIM;
    unsigned short* wV1 = wW1 + (size_t)H_DIM * H_DIM;
    unsigned short* wW2 = wV1 + (size_t)H_DIM * H_DIM;
    unsigned short* wV2 = wW2 + (size_t)H_DIM * H_DIM;

    const int threads = 256;
    {
        const int nx = B_ROWS * H_DIM;
        snn_prep_bf16<<<(nx + threads * 8 - 1) / (threads * 8), threads, 0, stream>>>(x, xbf, nx);
        const int nw = H_DIM * H_DIM;
        const int blk = (nw + threads * 8 - 1) / (threads * 8);
        snn_prep_bf16<<<blk, threads, 0, stream>>>(W1,  wW1, nw);
        snn_prep_bf16<<<blk, threads, 0, stream>>>(V1w, wV1, nw);
        snn_prep_bf16<<<blk, threads, 0, stream>>>(W2,  wW2, nw);
        snn_prep_bf16<<<blk, threads, 0, stream>>>(V2w, wV2, nw);
    }

    snn_persistent<<<B_ROWS / BT, 256, 0, stream>>>(
        xbf, wW1, wV1, wW2, wV2, b1, V1b, b2, V2b, W3, b3, out);
}
